// MotionDecoder_40029095198854
// MI455X (gfx1250) — matrix operand order repaired and checked by reading, not compiled, not run on hardware
//
#include <hip/hip_runtime.h>
#include <hip/hip_bf16.h>
#include <math.h>

// ---------------- constants from the reference ----------------
#define D_      256
#define H_      8
#define DH_     32
#define L_      4
#define DFF_    1024
#define KQ_     64      // intention queries
#define MOUT_   6
#define FLEN_   80
#define B_      64
#define A_      96
#define MM_     768

typedef __attribute__((ext_vector_type(16))) __bf16 v16bf;
typedef __attribute__((ext_vector_type(8)))  __bf16 v8bf;
typedef __attribute__((ext_vector_type(8)))  float  v8f;
typedef __attribute__((ext_vector_type(4)))  unsigned int v4u;
typedef __attribute__((ext_vector_type(8)))  int v8i;
typedef __attribute__((ext_vector_type(4)))  int v4i;

#if defined(__HIP_DEVICE_COMPILE__) && __has_builtin(__builtin_amdgcn_tensor_load_to_lds)
#define HAVE_TDM 1
#else
#define HAVE_TDM 0
#endif

// ---------------- helpers ----------------
__device__ __forceinline__ unsigned short f32_to_bf16_bits(float f) {
  union { float f; unsigned u; } v; v.f = f;
  unsigned u = v.u;
  u += 0x7FFFu + ((u >> 16) & 1u);   // round-to-nearest-even
  return (unsigned short)(u >> 16);
}
__device__ __forceinline__ unsigned pack_bf2(float a, float b) {
  return (unsigned)f32_to_bf16_bits(a) | ((unsigned)f32_to_bf16_bits(b) << 16);
}
__device__ __forceinline__ v16bf cat8(v8bf lo, v8bf hi) {
  return __builtin_shufflevector(lo, hi, 0,1,2,3,4,5,6,7,8,9,10,11,12,13,14,15);
}

#if HAVE_TDM
// ---- Tensor Data Mover: DMA a 2-D f32 tile (tileRows x tileCols) from global
// to LDS. Descriptor fields per CDNA5 ISA §8.3-8.5 (D# groups). Uniform args.
// This toolchain's builtin takes 6 args: (g0, g1, g2, g3, g_extra, cpol).
__device__ __forceinline__ void tdm_load_2d_f32(const float* gsrc, void* ldsDst,
                                                int tileCols, int tileRows,
                                                int rowStrideElems) {
  unsigned long long ga = (unsigned long long)(size_t)gsrc;
  unsigned lds = (unsigned)(size_t)ldsDst;       // low 32 bits = LDS byte address
  v4u g0;
  g0[0] = 1u;                                        // count=1, user descriptor
  g0[1] = lds;                                       // lds_addr
  g0[2] = (unsigned)(ga & 0xFFFFFFFFu);              // global_addr[31:0]
  g0[3] = (unsigned)((ga >> 32) & 0x1FFFFFFu)        // global_addr[56:32]
        | (2u << 30);                                // type = 2 ("image")
  unsigned td0 = (unsigned)rowStrideElems;           // tensor_dim0 (elements)
  unsigned td1 = (unsigned)tileRows;                 // tensor_dim1
  v8i g1;
  g1[0] = (int)(2u << 16);                           // wg_mask=0, data_size=4B
  g1[1] = (int)((td0 & 0xFFFFu) << 16);              // tensor_dim0[15:0] @ bits 63:48
  g1[2] = (int)(((td0 >> 16) & 0xFFFFu)              // tensor_dim0[31:16]
        | ((td1 & 0xFFFFu) << 16));                  // tensor_dim1[15:0]
  g1[3] = (int)(((td1 >> 16) & 0xFFFFu)              // tensor_dim1[31:16]
        | (((unsigned)tileCols & 0xFFFFu) << 16));   // tile_dim0
  g1[4] = (int)((unsigned)tileRows & 0xFFFFu);       // tile_dim1 (tile_dim2=0)
  g1[5] = (int)(unsigned)rowStrideElems;             // tensor_dim0_stride[31:0]
  g1[6] = 0;                                         // stride[47:32], dim1_stride lo
  g1[7] = 0;
  v4i gz4 = {0, 0, 0, 0};                            // groups 2/3 unused (<=2D)
  v8i gz8 = {0, 0, 0, 0, 0, 0, 0, 0};
  __builtin_amdgcn_tensor_load_to_lds(g0, g1, gz4, gz4, gz8, 0);
}
#endif

// =====================================================================
// Weight pre-pack: f32 (Kd x N) -> bf16 fragment order.
// Layout (ushort): [((kb*(Np/16) + s)*32 + L)*16 + j]
//   where lane L=(h<<4)|nl holds W[kb*32 + 16h + j][s*16 + nl], j=0..15.
// Zero-padded to Kp=ceil32(Kd), Np=ceil16(N).
// =====================================================================
__global__ void pack_w_kernel(const float* __restrict__ W, unsigned int* __restrict__ P,
                              int Kd, int N, int Kp, int Np)
{
  const int nsub = Np >> 4;
  const int totalPairs = (Kp >> 5) * nsub * 256;   // 512 ushorts per (kb,s) => 256 uints
  for (int e = blockIdx.x * blockDim.x + threadIdx.x; e < totalPairs;
       e += gridDim.x * blockDim.x) {
    int jp = e & 7;                 // uint pair within fragment: j0 = 2*jp
    int L  = (e >> 3) & 31;
    int bs = e >> 8;                // kb*nsub + s
    int s  = bs % nsub;
    int kb = bs / nsub;
    int h = L >> 4, nl = L & 15;
    int r0 = kb * 32 + 16 * h + 2 * jp;
    int c  = s * 16 + nl;
    float v0 = (r0 < Kd && c < N) ? W[(size_t)r0 * N + c] : 0.f;
    float v1 = (r0 + 1 < Kd && c < N) ? W[(size_t)(r0 + 1) * N + c] : 0.f;
    P[e] = pack_bf2(v0, v1);
  }
}

// =====================================================================
// GEMM: C = act(A @ W + bias). A: MxKd f32. W: packed bf16 (Kp x Np).
// Block = 128 threads (4 waves), tile 64(M) x 64(N), K-step 32.
// =====================================================================
#define TM 64
#define TN 64
#define TK 32
__global__ __launch_bounds__(128) void gemm_kernel(
    const float* __restrict__ A, const unsigned short* __restrict__ Wp,
    const float* __restrict__ bias, float* __restrict__ C,
    int M, int N, int Kd, int Kp, int Np, int relu)
{
  __shared__ alignas(16) unsigned short sA[TM][40];   // bf16 bits, stride 40 keeps 16B align
  const int tid  = threadIdx.x;
  const int wid  = tid >> 5;
  const int lane = tid & 31;
  const int h    = lane >> 4;
  const int nl   = lane & 15;
  const int row0 = blockIdx.y * TM;
  const int col0 = blockIdx.x * TN;
  const int nsub = Np >> 4;

  v8f acc0, acc1, acc2, acc3;
  const v8f vz = {0.f, 0.f, 0.f, 0.f, 0.f, 0.f, 0.f, 0.f};
  acc0 = vz; acc1 = vz; acc2 = vz; acc3 = vz;

  const bool fastA = (row0 + TM <= M) && ((Kd & 3) == 0);
  const bool fullN = (col0 + TN <= Np);
  const int  arow  = wid * 16 + nl;

  for (int kb = 0; kb * TK < Kp; ++kb) {
    const int k0 = kb * TK;
    // ---- stage A tile as bf16 ----
    if (fastA) {
      if (k0 + TK < Kd)   // prefetch next k-tile
        __builtin_prefetch(&A[(size_t)(row0 + (tid >> 1)) * Kd + k0 + TK + (tid & 1) * 16], 0, 3);
      for (int e = tid; e < TM * TK / 4; e += 128) {
        int r  = e >> 3;             // 8 float4 per row
        int c4 = (e & 7) * 4;
        int gc = k0 + c4;
        uint2 u; u.x = 0u; u.y = 0u;
        if (gc + 3 < Kd) {
          const float4 v = *(const float4*)&A[(size_t)(row0 + r) * Kd + gc];
          u.x = pack_bf2(v.x, v.y);
          u.y = pack_bf2(v.z, v.w);
        }
        *reinterpret_cast<uint2*>(&sA[r][c4]) = u;
      }
    } else {
      for (int i = tid; i < TM * TK; i += 128) {
        int r = i >> 5, c = i & 31;
        int gr = row0 + r, gc = k0 + c;
        float v = (gr < M && gc < Kd) ? A[(size_t)gr * Kd + gc] : 0.f;
        sA[r][c] = f32_to_bf16_bits(v);
      }
    }
    __syncthreads();

    // ---- A fragment: two aligned b128 LDS loads ----
    const v16bf af = cat8(*reinterpret_cast<const v8bf*>(&sA[arow][16 * h]),
                          *reinterpret_cast<const v8bf*>(&sA[arow][16 * h + 8]));

    // ---- B fragments from packed global (L2-resident) ----
    const unsigned short* base =
        Wp + (((size_t)kb * nsub + (col0 >> 4)) * 32 + lane) * 16;
    if (fullN) {
      // load all 8 b128s up front, then 4 unconditional WMMAs (accs stay pinned)
      const v16bf b0 = cat8(*reinterpret_cast<const v8bf*>(base + 0),
                            *reinterpret_cast<const v8bf*>(base + 8));
      const v16bf b1 = cat8(*reinterpret_cast<const v8bf*>(base + 512),
                            *reinterpret_cast<const v8bf*>(base + 520));
      const v16bf b2 = cat8(*reinterpret_cast<const v8bf*>(base + 1024),
                            *reinterpret_cast<const v8bf*>(base + 1032));
      const v16bf b3 = cat8(*reinterpret_cast<const v8bf*>(base + 1536),
                            *reinterpret_cast<const v8bf*>(base + 1544));
      acc0 = __builtin_amdgcn_wmma_f32_16x16x32_bf16(false, af, false, b0, (short)0, acc0, false, false);
      acc1 = __builtin_amdgcn_wmma_f32_16x16x32_bf16(false, af, false, b1, (short)0, acc1, false, false);
      acc2 = __builtin_amdgcn_wmma_f32_16x16x32_bf16(false, af, false, b2, (short)0, acc2, false, false);
      acc3 = __builtin_amdgcn_wmma_f32_16x16x32_bf16(false, af, false, b3, (short)0, acc3, false, false);
    } else {
      if (col0 < Np) {
        const v16bf b0 = cat8(*reinterpret_cast<const v8bf*>(base + 0),
                              *reinterpret_cast<const v8bf*>(base + 8));
        acc0 = __builtin_amdgcn_wmma_f32_16x16x32_bf16(false, af, false, b0, (short)0, acc0, false, false);
      }
      if (col0 + 16 < Np) {
        const v16bf b1 = cat8(*reinterpret_cast<const v8bf*>(base + 512),
                              *reinterpret_cast<const v8bf*>(base + 520));
        acc1 = __builtin_amdgcn_wmma_f32_16x16x32_bf16(false, af, false, b1, (short)0, acc1, false, false);
      }
      if (col0 + 32 < Np) {
        const v16bf b2 = cat8(*reinterpret_cast<const v8bf*>(base + 1024),
                              *reinterpret_cast<const v8bf*>(base + 1032));
        acc2 = __builtin_amdgcn_wmma_f32_16x16x32_bf16(false, af, false, b2, (short)0, acc2, false, false);
      }
      if (col0 + 48 < Np) {
        const v16bf b3 = cat8(*reinterpret_cast<const v8bf*>(base + 1536),
                              *reinterpret_cast<const v8bf*>(base + 1544));
        acc3 = __builtin_amdgcn_wmma_f32_16x16x32_bf16(false, af, false, b3, (short)0, acc3, false, false);
      }
    }
    __syncthreads();
  }

  // ---- epilogue: bias + act + store ----
  #pragma unroll
  for (int s = 0; s < 4; ++s) {
    const v8f& a = (s == 0) ? acc0 : (s == 1) ? acc1 : (s == 2) ? acc2 : acc3;
    int gcol = col0 + s * 16 + nl;
    if (gcol < N) {
      float bv = bias ? bias[gcol] : 0.f;
      #pragma unroll
      for (int r = 0; r < 8; ++r) {
        int grow = row0 + wid * 16 + r + 8 * h;
        if (grow < M) {
          float v = a[r] + bv;
          if (relu) v = v > 0.f ? v : 0.f;
          C[(size_t)grow * N + gcol] = v;
        }
      }
    }
  }
}

// =====================================================================
// Flash attention: one wave per (b, head, 16-query tile). kv chunks of 32.
// qh/kh/vh: (B, L, H*DH) f32. mask: (B, Lkv) bool8 or null.
// K/V chunks are DMA'd by the Tensor Data Mover into f32 staging buffers
// (double-buffered, overlapping WMMA+softmax), then converted to bf16
// fragment layouts in LDS.
// =====================================================================
__global__ __launch_bounds__(32) void attn_kernel(
    const float* __restrict__ qh, const float* __restrict__ kh,
    const float* __restrict__ vh, const unsigned char* __restrict__ mask,
    float* __restrict__ out, int Lq, int Lkv)
{
  __shared__ alignas(16) unsigned short sQ[16][40];
  __shared__ alignas(16) unsigned short sK[32][40];
  __shared__ alignas(16) unsigned short sVf[2][32][16];  // fragment-ready V
  __shared__ alignas(16) unsigned short sP[16][40];
#if HAVE_TDM
  __shared__ alignas(16) float sKst[2][32][32];          // TDM f32 staging
  __shared__ alignas(16) float sVst[2][32][32];
#endif

  const int nqt = Lq >> 4;
  const int bid = blockIdx.x;
  const int qt  = bid % nqt;
  const int hh  = (bid / nqt) % H_;
  const int b   = bid / (nqt * H_);
  const int q0  = qt * 16;

  const int lane = threadIdx.x;
  const int h    = lane >> 4;
  const int nl   = lane & 15;

  const float scale = 0.17677669529663687f;  // 1/sqrt(DH)

  // ---- stage Q tile (scaled) : lane (h,nl) fills row nl, cols 16h..16h+15 ----
  {
    const size_t rb = ((size_t)b * Lq + q0 + nl) * D_ + hh * DH_ + 16 * h;
    for (int c4 = 0; c4 < 16; c4 += 4) {
      const float4 v = *(const float4*)&qh[rb + c4];
      uint2 u; u.x = pack_bf2(v.x * scale, v.y * scale);
      u.y = pack_bf2(v.z * scale, v.w * scale);
      *reinterpret_cast<uint2*>(&sQ[nl][16 * h + c4]) = u;
    }
  }
  __syncthreads();

  const v16bf aq = cat8(*reinterpret_cast<const v8bf*>(&sQ[nl][16 * h]),
                        *reinterpret_cast<const v8bf*>(&sQ[nl][16 * h + 8]));

  const v8f vz = {0.f, 0.f, 0.f, 0.f, 0.f, 0.f, 0.f, 0.f};
  v8f o0 = vz, o1 = vz;
  float mstat[8], lstat[8];
  #pragma unroll
  for (int r = 0; r < 8; ++r) { mstat[r] = -3.0e38f; lstat[r] = 0.f; }

#if HAVE_TDM
  int cur = 0;
  // prime the pipeline: DMA chunk 0 of K and V
  tdm_load_2d_f32(&kh[((size_t)b * Lkv) * D_ + hh * DH_], &sKst[0][0][0], DH_, 32, D_);
  tdm_load_2d_f32(&vh[((size_t)b * Lkv) * D_ + hh * DH_], &sVst[0][0][0], DH_, 32, D_);
#endif

  for (int c = 0; c < Lkv; c += 32) {
#if HAVE_TDM
    __builtin_amdgcn_s_wait_tensorcnt(0);
    __syncthreads();
    // ---- convert staged f32 -> bf16 layouts: one kv row per lane ----
    {
      const int r = lane;
      #pragma unroll
      for (int c4 = 0; c4 < 32; c4 += 4) {
        const float4 kv = *(const float4*)&sKst[cur][r][c4];
        uint2 u; u.x = pack_bf2(kv.x, kv.y); u.y = pack_bf2(kv.z, kv.w);
        *reinterpret_cast<uint2*>(&sK[r][c4]) = u;
        const float4 vv = *(const float4*)&sVst[cur][r][c4];
        const float ve[4] = {vv.x, vv.y, vv.z, vv.w};
        #pragma unroll
        for (int t = 0; t < 4; ++t) {
          int cc = c4 + t;
          sVf[cc >> 4][((r >> 4) << 4) | (cc & 15)][r & 15] = f32_to_bf16_bits(ve[t]);
        }
      }
    }
    __syncthreads();
    // ---- kick off DMA of the next chunk into the other buffer ----
    if (c + 32 < Lkv) {
      tdm_load_2d_f32(&kh[((size_t)b * Lkv + c + 32) * D_ + hh * DH_],
                      &sKst[cur ^ 1][0][0], DH_, 32, D_);
      tdm_load_2d_f32(&vh[((size_t)b * Lkv + c + 32) * D_ + hh * DH_],
                      &sVst[cur ^ 1][0][0], DH_, 32, D_);
    }
#else
    // ---- direct path: stage K (row-major) and V (fragment-ready) ----
    {
      const int r = lane;
      const size_t rb = ((size_t)b * Lkv + c + r) * D_ + hh * DH_;
      #pragma unroll
      for (int c4 = 0; c4 < 32; c4 += 4) {
        const float4 kv = *(const float4*)&kh[rb + c4];
        uint2 u; u.x = pack_bf2(kv.x, kv.y); u.y = pack_bf2(kv.z, kv.w);
        *reinterpret_cast<uint2*>(&sK[r][c4]) = u;
        const float4 vv = *(const float4*)&vh[rb + c4];
        const float ve[4] = {vv.x, vv.y, vv.z, vv.w};
        #pragma unroll
        for (int t = 0; t < 4; ++t) {
          int cc = c4 + t;
          sVf[cc >> 4][((r >> 4) << 4) | (cc & 15)][r & 15] = f32_to_bf16_bits(ve[t]);
        }
      }
    }
    __syncthreads();
#endif

    // ---- S = Q @ K^T : two 16x16 subtiles (B fragment contiguous along DH) ----
    v8f s0 = vz, s1 = vz;
    {
      const v16bf bk0 = cat8(*reinterpret_cast<const v8bf*>(&sK[nl][16 * h]),
                             *reinterpret_cast<const v8bf*>(&sK[nl][16 * h + 8]));
      const v16bf bk1 = cat8(*reinterpret_cast<const v8bf*>(&sK[16 + nl][16 * h]),
                             *reinterpret_cast<const v8bf*>(&sK[16 + nl][16 * h + 8]));
      s0 = __builtin_amdgcn_wmma_f32_16x16x32_bf16(false, aq, false, bk0, (short)0, s0, false, false);
      s1 = __builtin_amdgcn_wmma_f32_16x16x32_bf16(false, aq, false, bk1, (short)0, s1, false, false);
    }

    bool mv0 = true, mv1 = true;
    if (mask) {
      mv0 = mask[(size_t)b * Lkv + c + nl] != 0;
      mv1 = mask[(size_t)b * Lkv + c + 16 + nl] != 0;
    }

    // ---- online softmax (rows r+8h live on this half's 16 lanes) ----
    #pragma unroll
    for (int r = 0; r < 8; ++r) {
      float sv0 = mv0 ? s0[r] : -1.0e9f;
      float sv1 = mv1 ? s1[r] : -1.0e9f;
      float cm = fmaxf(sv0, sv1);
      cm = fmaxf(cm, __shfl_xor(cm, 1, 32));
      cm = fmaxf(cm, __shfl_xor(cm, 2, 32));
      cm = fmaxf(cm, __shfl_xor(cm, 4, 32));
      cm = fmaxf(cm, __shfl_xor(cm, 8, 32));
      float newm = fmaxf(mstat[r], cm);
      float alpha = __expf(mstat[r] - newm);
      float p0 = __expf(sv0 - newm);
      float p1 = __expf(sv1 - newm);
      float ps = p0 + p1;
      ps += __shfl_xor(ps, 1, 32);
      ps += __shfl_xor(ps, 2, 32);
      ps += __shfl_xor(ps, 4, 32);
      ps += __shfl_xor(ps, 8, 32);
      lstat[r] = lstat[r] * alpha + ps;
      mstat[r] = newm;
      o0[r] *= alpha;
      o1[r] *= alpha;
      sP[r + 8 * h][nl]      = f32_to_bf16_bits(p0);
      sP[r + 8 * h][16 + nl] = f32_to_bf16_bits(p1);
    }
    __syncthreads();

    // ---- O += P @ V ----
    const v16bf ap = cat8(*reinterpret_cast<const v8bf*>(&sP[nl][16 * h]),
                          *reinterpret_cast<const v8bf*>(&sP[nl][16 * h + 8]));
    {
      const v16bf bv0 = cat8(*reinterpret_cast<const v8bf*>(&sVf[0][lane][0]),
                             *reinterpret_cast<const v8bf*>(&sVf[0][lane][8]));
      const v16bf bv1 = cat8(*reinterpret_cast<const v8bf*>(&sVf[1][lane][0]),
                             *reinterpret_cast<const v8bf*>(&sVf[1][lane][8]));
      o0 = __builtin_amdgcn_wmma_f32_16x16x32_bf16(false, ap, false, bv0, (short)0, o0, false, false);
      o1 = __builtin_amdgcn_wmma_f32_16x16x32_bf16(false, ap, false, bv1, (short)0, o1, false, false);
    }
    __syncthreads();
#if HAVE_TDM
    cur ^= 1;
#endif
  }

  #pragma unroll
  for (int r = 0; r < 8; ++r) {
    float inv = 1.0f / lstat[r];
    int qrow = q0 + r + 8 * h;
    size_t base = ((size_t)b * Lq + qrow) * D_ + hh * DH_;
    out[base + nl]      = o0[r] * inv;
    out[base + 16 + nl] = o1[r] * inv;
  }
}

// ---------------- x = LN(x + delta) * g + b, one wave per 256-wide row ----------------
__global__ __launch_bounds__(32) void ln_residual_kernel(
    float* __restrict__ x, const float* __restrict__ delta,
    const float* __restrict__ g, const float* __restrict__ bln)
{
  const int row = blockIdx.x;
  const int lane = threadIdx.x;
  float v[8];
  float s = 0.f;
  #pragma unroll
  for (int i = 0; i < 8; ++i) {
    int c = lane * 8 + i;
    v[i] = x[(size_t)row * D_ + c] + delta[(size_t)row * D_ + c];
    s += v[i];
  }
  s += __shfl_xor(s, 1, 32);  s += __shfl_xor(s, 2, 32);
  s += __shfl_xor(s, 4, 32);  s += __shfl_xor(s, 8, 32);
  s += __shfl_xor(s, 16, 32);
  float mean = s * (1.0f / D_);
  float vs = 0.f;
  #pragma unroll
  for (int i = 0; i < 8; ++i) { float d = v[i] - mean; vs += d * d; }
  vs += __shfl_xor(vs, 1, 32);  vs += __shfl_xor(vs, 2, 32);
  vs += __shfl_xor(vs, 4, 32);  vs += __shfl_xor(vs, 8, 32);
  vs += __shfl_xor(vs, 16, 32);
  float rs = rsqrtf(vs * (1.0f / D_) + 1e-5f);
  #pragma unroll
  for (int i = 0; i < 8; ++i) {
    int c = lane * 8 + i;
    x[(size_t)row * D_ + c] = (v[i] - mean) * rs * g[c] + bln[c];
  }
}

// ---------------- q[b,k,:] = intent[k,:] + tctx[b,:] ----------------
__global__ void build_q_kernel(const float* __restrict__ intent,
                               const float* __restrict__ tctx,
                               float* __restrict__ q)
{
  int i = blockIdx.x * 256 + threadIdx.x;     // B*K*D = 1048576 total
  int d  = i & (D_ - 1);
  int kq = (i >> 8) & (KQ_ - 1);
  int b  = i >> 14;
  q[i] = intent[kq * D_ + d] + tctx[b * D_ + d];
}

// ---------------- greedy NMS, one block per batch ----------------
__global__ __launch_bounds__(64) void nms_kernel(
    const float* __restrict__ trajs,   // (B, K, FLEN*2)
    const float* __restrict__ scores,  // (B, K)
    float* __restrict__ sel_trajs,     // (B, MOUT, FLEN*2)
    float* __restrict__ sel_scores,    // (B, MOUT)
    int* __restrict__ sel_idx)         // (B, MOUT)
{
  const int b = blockIdx.x;
  const int t = threadIdx.x;
  __shared__ float ex[KQ_], ey[KQ_];
  __shared__ int   sup[KQ_];
  __shared__ float red_s[KQ_];
  __shared__ int   red_i[KQ_];
  __shared__ int   pick[MOUT_];
  __shared__ int   pvalid[MOUT_];

  const float* tb = trajs + (size_t)b * KQ_ * (FLEN_ * 2);
  ex[t] = tb[t * (FLEN_ * 2) + (FLEN_ - 1) * 2 + 0];
  ey[t] = tb[t * (FLEN_ * 2) + (FLEN_ - 1) * 2 + 1];
  sup[t] = 0;
  __syncthreads();

  for (int m = 0; m < MOUT_; ++m) {
    red_s[t] = sup[t] ? -3.0e38f : scores[(size_t)b * KQ_ + t];
    red_i[t] = t;
    __syncthreads();
    for (int off = 32; off > 0; off >>= 1) {
      if (t < off) {
        if (red_s[t + off] > red_s[t]) {   // strict > keeps lowest index on ties
          red_s[t] = red_s[t + off];
          red_i[t] = red_i[t + off];
        }
      }
      __syncthreads();
    }
    if (t == 0) {
      pick[m]   = red_i[0];
      pvalid[m] = (red_s[0] > -2.9e38f) ? 1 : 0;
    }
    __syncthreads();
    int i = pick[m];
    float dx = ex[t] - ex[i], dy = ey[t] - ey[i];
    if (sqrtf(dx * dx + dy * dy) < 2.0f) sup[t] = 1;
    __syncthreads();
  }

  for (int m = 0; m < MOUT_; ++m) {
    int idx = pick[m], val = pvalid[m];
    for (int e = t; e < FLEN_ * 2; e += 64)
      sel_trajs[((size_t)b * MOUT_ + m) * (FLEN_ * 2) + e] = val ? tb[idx * (FLEN_ * 2) + e] : 0.f;
    if (t == 0) {
      sel_scores[b * MOUT_ + m] = val ? scores[(size_t)b * KQ_ + idx] : -1.0e9f;
      sel_idx[b * MOUT_ + m]    = val ? idx : 0;
    }
  }
}

// ---------------- host orchestration ----------------
struct PW   { const unsigned short* p; int Kp, Np; };
struct LinP { PW W; const float* b; };
struct AttnP { LinP k, o, q, v; };                 // jax pytree sorted: k,o,q,v
struct LayerP {
  AttnP ca_a, ca_m; LinP ff1, ff2;
  const float *ln1b, *ln1g, *ln2b, *ln2g, *ln3b, *ln3g, *ln4b, *ln4g;
  AttnP sa;
};

static inline void gemm(hipStream_t st, const float* A, const PW& w,
                        const float* bias, float* C, int M, int N, int Kd, int relu) {
  dim3 grid((N + TN - 1) / TN, (M + TM - 1) / TM);
  gemm_kernel<<<grid, dim3(128), 0, st>>>(A, w.p, bias, C, M, N, Kd, w.Kp, w.Np, relu);
}

extern "C" void kernel_launch(void* const* d_in, const int* in_sizes, int n_in,
                              void* d_out, int out_size, void* d_ws, size_t ws_size,
                              hipStream_t stream) {
  (void)in_sizes; (void)n_in; (void)out_size; (void)ws_size;

  const float* tat   = (const float*)d_in[0];            // (B, D)
  const float* atoks = (const float*)d_in[1];            // (B, A, D)
  const float* mtoks = (const float*)d_in[2];            // (B, MM, D)
  const unsigned char* amask = (const unsigned char*)d_in[3];  // (B, A) bool8
  const unsigned char* mmask = (const unsigned char*)d_in[4];  // (B, MM) bool8
  const float* ipts  = (const float*)d_in[5];            // (K, 2)

  // ---- workspace arena (floats), then packed-weight arena (ushort) ----
  float* ws  = (float*)d_ws;
  float* q   = ws;                               // B*K*D   = 1,048,576
  float* qh  = q   + (size_t)B_ * KQ_ * D_;
  float* kh  = qh  + (size_t)B_ * KQ_ * D_;      // B*MM*D = 12,582,912
  float* vh  = kh  + (size_t)B_ * MM_ * D_;
  float* att = vh  + (size_t)B_ * MM_ * D_;
  float* ffh = att + (size_t)B_ * KQ_ * D_;      // B*K*DFF
  float* sm1 = ffh + (size_t)B_ * KQ_ * DFF_;    // intent
  float* sm2 = sm1 + KQ_ * D_;                   // tctx
  float* sm3 = sm2 + B_ * D_;                    // tmp
  float* sm4 = sm3 + KQ_ * D_;                   // tmp
  unsigned short* wcur = (unsigned short*)(sm4 + B_ * D_);

  auto packW = [&](const float* W, int Kd, int N) -> PW {
    int Kp = (Kd + 31) & ~31, Np = (N + 15) & ~15;
    int totalPairs = (Kp >> 5) * (Np >> 4) * 256;
    unsigned short* dst = wcur;
    wcur += (size_t)totalPairs * 2;
    int blocks = (totalPairs + 255) / 256;
    if (blocks > 2048) blocks = 2048;
    pack_w_kernel<<<dim3(blocks), dim3(256), 0, stream>>>(
        W, (unsigned int*)dst, Kd, N, Kp, Np);
    return PW{ dst, Kp, Np };
  };

  // ---- params as jax pytree leaves: sorted dict keys, lists in order ----
  int idx = 6;
  auto nf = [&]() { return (const float*)d_in[idx++]; };
  auto rdLin = [&](int Kd, int N, bool doPack) -> LinP {
    const float* W = nf(); const float* b = nf();
    LinP r; r.b = b;
    r.W = doPack ? packW(W, Kd, N) : PW{ nullptr, 0, 0 };
    return r;
  };

  // heads: sorted s1,s2,t1,t2 — only last layer's head is ever used
  LinP h_s1{}, h_s2{}, h_t1{}, h_t2{};
  for (int i = 0; i < L_; ++i) {
    bool last = (i == L_ - 1);
    LinP s1 = rdLin(D_, D_ / 2, last);
    LinP s2 = rdLin(D_ / 2, 1, last);
    LinP t1 = rdLin(D_, D_, last);
    LinP t2 = rdLin(D_, FLEN_ * 2, last);
    if (last) { h_s1 = s1; h_s2 = s2; h_t1 = t1; h_t2 = t2; }
  }
  LinP ie1 = rdLin(2, D_, true);
  LinP ie2 = rdLin(D_, D_, true);
  LayerP lp[L_];
  for (int i = 0; i < L_; ++i) {
    auto rdA = [&](AttnP& a) {
      a.k = rdLin(D_, D_, true); a.o = rdLin(D_, D_, true);
      a.q = rdLin(D_, D_, true); a.v = rdLin(D_, D_, true);
    };
    rdA(lp[i].ca_a);
    rdA(lp[i].ca_m);
    lp[i].ff1 = rdLin(D_, DFF_, true);
    lp[i].ff2 = rdLin(DFF_, D_, true);
    lp[i].ln1b = nf(); lp[i].ln1g = nf();
    lp[i].ln2b = nf(); lp[i].ln2g = nf();
    lp[i].ln3b = nf(); lp[i].ln3g = nf();
    lp[i].ln4b = nf(); lp[i].ln4g = nf();
    rdA(lp[i].sa);
  }
  LinP tc1 = rdLin(D_, D_, true);
  LinP tc2 = rdLin(D_, D_, true);

  // ---- output regions (floats): sel_trajs, sel_scores, sel_idx, trajs, scores ----
  float* outf       = (float*)d_out;
  float* sel_trajs  = outf;                         // 64*6*160
  float* sel_scores = outf + 61440;                 // 384
  int*   sel_idxp   = (int*)(outf + 61824);         // 384
  float* trajs      = outf + 62208;                 // 64*64*160
  float* scores     = outf + 717568;                // 4096

  const int NQ = B_ * KQ_;                          // 4096
  const dim3 attnGrid(B_ * H_ * (KQ_ / 16));

  // intent / target-context embeddings
  gemm(stream, ipts, ie1.W, ie1.b, sm3, KQ_, D_, 2, 1);
  gemm(stream, sm3,  ie2.W, ie2.b, sm1, KQ_, D_, D_, 0);
  gemm(stream, tat,  tc1.W, tc1.b, sm4, B_,  D_, D_, 1);
  gemm(stream, sm4,  tc2.W, tc2.b, sm2, B_,  D_, D_, 0);
  build_q_kernel<<<dim3(NQ), dim3(256), 0, stream>>>(sm1, sm2, q);

  for (int l = 0; l < L_; ++l) {
    const LayerP& P = lp[l];
    // --- self attention ---
    gemm(stream, q, P.sa.q.W, P.sa.q.b, qh, NQ, D_, D_, 0);
    gemm(stream, q, P.sa.k.W, P.sa.k.b, kh, NQ, D_, D_, 0);
    gemm(stream, q, P.sa.v.W, P.sa.v.b, vh, NQ, D_, D_, 0);
    attn_kernel<<<attnGrid, dim3(32), 0, stream>>>(qh, kh, vh, nullptr, att, KQ_, KQ_);
    gemm(stream, att, P.sa.o.W, P.sa.o.b, qh, NQ, D_, D_, 0);
    ln_residual_kernel<<<dim3(NQ), dim3(32), 0, stream>>>(q, qh, P.ln1g, P.ln1b);
    // --- cross attention: agents ---
    gemm(stream, q,     P.ca_a.q.W, P.ca_a.q.b, qh, NQ,      D_, D_, 0);
    gemm(stream, atoks, P.ca_a.k.W, P.ca_a.k.b, kh, B_ * A_, D_, D_, 0);
    gemm(stream, atoks, P.ca_a.v.W, P.ca_a.v.b, vh, B_ * A_, D_, D_, 0);
    attn_kernel<<<attnGrid, dim3(32), 0, stream>>>(qh, kh, vh, amask, att, KQ_, A_);
    gemm(stream, att, P.ca_a.o.W, P.ca_a.o.b, qh, NQ, D_, D_, 0);
    ln_residual_kernel<<<dim3(NQ), dim3(32), 0, stream>>>(q, qh, P.ln2g, P.ln2b);
    // --- cross attention: map ---
    gemm(stream, q,     P.ca_m.q.W, P.ca_m.q.b, qh, NQ,       D_, D_, 0);
    gemm(stream, mtoks, P.ca_m.k.W, P.ca_m.k.b, kh, B_ * MM_, D_, D_, 0);
    gemm(stream, mtoks, P.ca_m.v.W, P.ca_m.v.b, vh, B_ * MM_, D_, D_, 0);
    attn_kernel<<<attnGrid, dim3(32), 0, stream>>>(qh, kh, vh, mmask, att, KQ_, MM_);
    gemm(stream, att, P.ca_m.o.W, P.ca_m.o.b, qh, NQ, D_, D_, 0);
    ln_residual_kernel<<<dim3(NQ), dim3(32), 0, stream>>>(q, qh, P.ln3g, P.ln3b);
    // --- FFN ---
    gemm(stream, q,   P.ff1.W, P.ff1.b, ffh, NQ, DFF_, D_, 1);
    gemm(stream, ffh, P.ff2.W, P.ff2.b, qh,  NQ, D_,  DFF_, 0);
    ln_residual_kernel<<<dim3(NQ), dim3(32), 0, stream>>>(q, qh, P.ln4g, P.ln4b);
  }

  // heads (only the last layer's head output survives in the reference)
  gemm(stream, q,   h_t1.W, h_t1.b, att,    NQ, D_,        D_, 1);
  gemm(stream, att, h_t2.W, h_t2.b, trajs,  NQ, FLEN_ * 2, D_, 0);
  gemm(stream, q,   h_s1.W, h_s1.b, qh,     NQ, D_ / 2,    D_, 1);
  gemm(stream, qh,  h_s2.W, h_s2.b, scores, NQ, 1,     D_ / 2, 0);

  nms_kernel<<<dim3(B_), dim3(64), 0, stream>>>(trajs, scores, sel_trajs, sel_scores, sel_idxp);
}
